// Beamformer_Network_8873402433746
// MI455X (gfx1250) — matrix operand order repaired and checked by reading, not compiled, not run on hardware
//
#include <hip/hip_runtime.h>
#include <hip/hip_bf16.h>
#include <math.h>

// ---------------- problem constants ----------------
constexpr int kB      = 1024;  // batch
constexpr int kSC     = 64;    // subcarriers
constexpr int kGAP    = 8;
constexpr int kPS     = 8;     // pilot subcarriers
constexpr int kPN     = 8;     // pilot slots
constexpr int kUE     = 16;    // UE antennas
constexpr int kBS     = 64;    // BS antennas
constexpr int kRF     = 4;
constexpr int kDS     = 2;
constexpr int kFEAT   = 256;
constexpr int kK      = 256;   // VQ codebook size

typedef _Float16 h16;
typedef _Float16 v16h __attribute__((ext_vector_type(16)));
typedef _Float16 v8h  __attribute__((ext_vector_type(8)));
typedef float    v8f  __attribute__((ext_vector_type(8)));

__device__ __forceinline__ float mishf(float x) {
  return x * tanhf(log1pf(expf(x)));
}

// ============================================================
// K0: zero the two loss slots at the tail of d_out
// ============================================================
__global__ void zero_loss_kernel(float* p) {
  if (threadIdx.x < 2) p[threadIdx.x] = 0.0f;
}

// ============================================================
// K1: pilot constants.
// fvec[pn,ps,k] = (F[k,0]+F[k,1])/||F||_F  (complex, k in [0,64))
// Wc[pn,ps,d,u] = W[d,u]/||W||_F * sqrt(2) (complex)
// grid: 64 blocks (pn*8+ps), 64 threads
// ============================================================
__global__ void pilot_const_kernel(const float* __restrict__ WRFt,
                                   const float* __restrict__ FRFt,
                                   const float* __restrict__ WBBr,
                                   const float* __restrict__ WBBi,
                                   const float* __restrict__ FBBr,
                                   const float* __restrict__ FBBi,
                                   float* fvr, float* fvi,
                                   float* Wcr, float* Wci) {
  int pnps = blockIdx.x;
  int pn = pnps >> 3;
  int t = threadIdx.x;  // 0..63
  __shared__ float red[64];

  // ---- F = FRF @ FBB, row bs = t ----
  float fr0 = 0.f, fi0 = 0.f, fr1 = 0.f, fi1 = 0.f;
  {
    int bs = t;
    for (int r = 0; r < kRF; ++r) {
      float th = FRFt[pn * kBS * kRF + bs * kRF + r];
      float ar = cosf(th) * 0.125f;             // 1/sqrt(64)
      float ai = sinf(th) * 0.125f;
      int fb = ((pnps) * kRF + r) * kDS;
      float br0 = FBBr[fb + 0], bi0 = FBBi[fb + 0];
      float br1 = FBBr[fb + 1], bi1 = FBBi[fb + 1];
      fr0 += ar * br0 - ai * bi0;  fi0 += ar * bi0 + ai * br0;
      fr1 += ar * br1 - ai * bi1;  fi1 += ar * bi1 + ai * br1;
    }
  }
  red[t] = fr0 * fr0 + fi0 * fi0 + fr1 * fr1 + fi1 * fi1;
  __syncthreads();
  for (int s = 32; s > 0; s >>= 1) { if (t < s) red[t] += red[t + s]; __syncthreads(); }
  {
    float invp = 1.0f / sqrtf(red[0]);
    fvr[pnps * kBS + t] = (fr0 + fr1) * invp;
    fvi[pnps * kBS + t] = (fi0 + fi1) * invp;
  }
  __syncthreads();

  // ---- W = WBB @ WRF, entry (d,u) for t < 32 ----
  float wr = 0.f, wi = 0.f;
  {
    int d = t >> 4, u = t & 15;
    if (t < 32) {
      for (int r = 0; r < kRF; ++r) {
        float th = WRFt[pn * kRF * kUE + r * kUE + u];
        float cr = cosf(th) * 0.25f;            // 1/sqrt(16)
        float ci = sinf(th) * 0.25f;
        int wb = ((pnps) * kDS + d) * kRF + r;
        float br = WBBr[wb], bi = WBBi[wb];
        wr += br * cr - bi * ci;
        wi += br * ci + bi * cr;
      }
    }
  }
  red[t] = (t < 32) ? (wr * wr + wi * wi) : 0.0f;
  __syncthreads();
  for (int s = 32; s > 0; s >>= 1) { if (t < s) red[t] += red[t + s]; __syncthreads(); }
  if (t < 32) {
    float sca = 1.41421356237f / sqrtf(red[0]);  // sqrt(DS)/||W||
    Wcr[pnps * 32 + t] = wr * sca;               // layout [pnps][d*16+u]
    Wci[pnps * 32 + t] = wi * sca;
  }
}

// ============================================================
// K2: pilot projection -> fb (B,256)
// grid: B*PS blocks, 128 threads
// ============================================================
__global__ void pilot_proj_kernel(const float* __restrict__ Hr,
                                  const float* __restrict__ Hi,
                                  const float* __restrict__ fvr,
                                  const float* __restrict__ fvi,
                                  const float* __restrict__ Wcr,
                                  const float* __restrict__ Wci,
                                  float* __restrict__ fb) {
  int blk = blockIdx.x;
  int b = blk >> 3, ps = blk & 7;
  int sc = ps * kGAP;
  int t = threadIdx.x;  // 128
  __shared__ float sHr[kUE * kBS], sHi[kUE * kBS];
  __shared__ float sfr[kPN * kBS], sfi[kPN * kBS];
  __shared__ float gr[kPN * kUE], gi[kPN * kUE];

  size_t hb = ((size_t)(b * kSC + sc)) * kUE * kBS;
  for (int idx = t; idx < kUE * kBS; idx += 128) {
    sHr[idx] = Hr[hb + idx];
    sHi[idx] = Hi[hb + idx];
  }
  for (int idx = t; idx < kPN * kBS; idx += 128) {
    int pn = idx >> 6, k = idx & 63;
    sfr[idx] = fvr[(pn * 8 + ps) * kBS + k];
    sfi[idx] = fvi[(pn * 8 + ps) * kBS + k];
  }
  __syncthreads();
  {
    int pn = t >> 4, u = t & 15;
    const float* hr = &sHr[u * kBS];
    const float* hi = &sHi[u * kBS];
    const float* fr = &sfr[pn * kBS];
    const float* fi = &sfi[pn * kBS];
    float ar = 0.f, ai = 0.f;
    for (int k = 0; k < kBS; ++k) {
      float a = hr[k], bq = hi[k], c = fr[k], d = fi[k];
      ar += a * c - bq * d;
      ai += a * d + bq * c;
    }
    gr[pn * kUE + u] = ar;
    gi[pn * kUE + u] = ai;
  }
  __syncthreads();
  if (t < 32) {
    int pn = t >> 2, d = (t >> 1) & 1, part = t & 1;
    const float* wr = &Wcr[(pn * 8 + ps) * 32 + d * 16];
    const float* wi = &Wci[(pn * 8 + ps) * 32 + d * 16];
    const float* grp = &gr[pn * kUE];
    const float* gip = &gi[pn * kUE];
    float v = 0.f;
    for (int u = 0; u < kUE; ++u) {
      v += part ? (wr[u] * gip[u] + wi[u] * grp[u])
                : (wr[u] * grp[u] - wi[u] * gip[u]);
    }
    fb[(size_t)b * kFEAT + part * 128 + pn * 16 + ps * 2 + d] = v;
  }
}

// ============================================================
// K3: VQ quantization -> fbq (f16), losses via atomicAdd
// grid: B blocks, 128 threads
// ============================================================
__global__ void vq_kernel(const float* __restrict__ fb,
                          const float* __restrict__ embed,
                          h16* __restrict__ fbq16,
                          float* __restrict__ lossbuf) {
  int b = blockIdx.x, t = threadIdx.x;  // 128
  __shared__ float sfb[kFEAT];
  __shared__ float semb[kK * 2];
  __shared__ float red[128];
  __shared__ float sq_s;

  sfb[t]       = fb[(size_t)b * kFEAT + t];
  sfb[t + 128] = fb[(size_t)b * kFEAT + t + 128];
  for (int i = t; i < kK * 2; i += 128) semb[i] = embed[i];
  __syncthreads();

  red[t] = sfb[t] * sfb[t] + sfb[t + 128] * sfb[t + 128];
  __syncthreads();
  for (int s = 64; s > 0; s >>= 1) { if (t < s) red[t] += red[t + s]; __syncthreads(); }
  float scale = sqrtf(red[0]);
  if (t == 0) {
    float c = fminf(fmaxf(scale + 1.0f, 1.0f), 16.0f);
    float lx = logf(c) * 0.721347520444f;  // 1/log(4)
    float ssum = 0.0f;
    for (int j = 0; j < 8; ++j) {
      int bj = ((int)(lx * (float)(1 << j))) & 1;
      ssum += (float)bj / (float)(1 << j);
    }
    sq_s = exp2f(2.0f * ssum) - 1.0f;
  }
  __syncthreads();
  float sq = sq_s;
  float inv = 1.0f / scale;
  float fx = sfb[2 * t] * inv, fy = sfb[2 * t + 1] * inv;
  int best = 0;
  float bd = 3.0e38f;
  for (int k = 0; k < kK; ++k) {
    float dx = fx - semb[2 * k], dy = fy - semb[2 * k + 1];
    float dd = dx * dx + dy * dy;
    if (dd < bd) { bd = dd; best = k; }
  }
  float ex = semb[2 * best], ey = semb[2 * best + 1];
  __syncthreads();
  red[t] = bd;  // (ex-fx)^2 + (ey-fy)^2
  __syncthreads();
  for (int s = 64; s > 0; s >>= 1) { if (t < s) red[t] += red[t + s]; __syncthreads(); }
  if (t == 0) {
    float lc = red[0] * (1.0f / (float)(kB * 128 * 2));  // mean over (B,M,2)
    atomicAdd(&lossbuf[0], lc);
    atomicAdd(&lossbuf[1], lc);
  }
  fbq16[(size_t)b * kFEAT + 2 * t]     = (h16)(ex * sq);
  fbq16[(size_t)b * kFEAT + 2 * t + 1] = (h16)(ey * sq);
}

// ============================================================
// K4: f32 -> f16 transpose-convert: src (K,N) f32 -> dst (N,K) f16
// 32x32 LDS tiles; grid (N/32, K/32), 256 threads
// ============================================================
__global__ __launch_bounds__(256) void cvt_t_kernel(const float* __restrict__ src,
                                                    h16* __restrict__ dst,
                                                    int K, int N) {
  __shared__ float tile[32][33];
  int kb = blockIdx.y * 32, nb = blockIdx.x * 32;
  int tx = threadIdx.x & 31, ty = threadIdx.x >> 5;  // ty in [0,8)
  for (int j = 0; j < 4; ++j)
    tile[ty + j * 8][tx] = src[(size_t)(kb + ty + j * 8) * N + nb + tx];
  __syncthreads();
  for (int j = 0; j < 4; ++j)
    dst[(size_t)(nb + ty + j * 8) * K + kb + tx] = (h16)tile[tx][ty + j * 8];
}

// ============================================================
// K5: WMMA GEMM  out = act(A(MxK,f16) @ W + bias), W given
//     PRE-TRANSPOSED as Wt (N,K) f16.
// block: 256 threads (8 waves), tile 256(M)x64(N), BK=32,
// double-buffered LDS; each wave: 64x32 = 4x2 WMMA tiles
// ============================================================
__global__ __launch_bounds__(256) void gemm_wmma_kernel(
    const h16* __restrict__ A, const h16* __restrict__ Wt,
    const float* __restrict__ bias, float* __restrict__ outF,
    h16* __restrict__ outH, int M, int N, int K, int act) {
  __shared__ __align__(16) h16 sA[2][256 * 32];  // [m][k]
  __shared__ __align__(16) h16 sB[2][64 * 32];   // [n][k]
  int t = threadIdx.x;
  int lane = t & 31, w = t >> 5;
  int m0 = blockIdx.y * 256, n0 = blockIdx.x * 64;
  int mw = (w >> 1) * 64, nw = (w & 1) * 32;

  // global chunk assignment (16-byte chunks, coalesced across threads)
  int ar[4], ac[4];
  for (int j = 0; j < 4; ++j) {
    int c = t + j * 256;           // 0..1023 -> A 256x32
    ar[j] = c >> 2;
    ac[j] = (c & 3) * 8;
  }
  int br = t >> 2, bc = (t & 3) * 8;  // 0..255 -> B 64x32

  const int nK = K >> 5;
  v8f acc[4][2] = {};
  v8h aReg[4], bReg;

  // prologue: stage tile 0
  for (int j = 0; j < 4; ++j)
    aReg[j] = *(const v8h*)&A[(size_t)(m0 + ar[j]) * K + ac[j]];
  bReg = *(const v8h*)&Wt[(size_t)(n0 + br) * K + bc];
  for (int j = 0; j < 4; ++j)
    *(v8h*)&sA[0][ar[j] * 32 + ac[j]] = aReg[j];
  *(v8h*)&sB[0][br * 32 + bc] = bReg;
  __syncthreads();

  for (int kt = 0; kt < nK; ++kt) {
    int cur = kt & 1, nxt = cur ^ 1;
    int kg = (kt + 1) << 5;
    // 1) issue next tile's global loads (latency hidden behind WMMAs)
    if (kt + 1 < nK) {
      for (int j = 0; j < 4; ++j)
        aReg[j] = *(const v8h*)&A[(size_t)(m0 + ar[j]) * K + kg + ac[j]];
      bReg = *(const v8h*)&Wt[(size_t)(n0 + br) * K + kg + bc];
    }
    if (kt + 2 < nK)  // L2 prefetch two tiles ahead -> global_prefetch_b8
      __builtin_prefetch(&A[(size_t)(m0 + ar[0]) * K + ((kt + 2) << 5) + ac[0]], 0, 1);

    // 2) fragments + WMMA from current buffer
    // A 16x32 f16: lane<16 -> M=lane, halves {K0..7, K16..23};
    //              lane>=16 -> same M, halves {K8..15, K24..31}
    v16h af[4], bf[2];
    int mrow = lane & 15, kb8 = (lane >> 4) * 8;
    for (int mi = 0; mi < 4; ++mi) {
      const h16* p = &sA[cur][(mw + mi * 16 + mrow) * 32];
      v8h lo = *(const v8h*)&p[kb8];
      v8h hi = *(const v8h*)&p[kb8 + 16];
      af[mi] = __builtin_shufflevector(lo, hi, 0, 1, 2, 3, 4, 5, 6, 7,
                                       8, 9, 10, 11, 12, 13, 14, 15);
    }
    // B 32x16 f16: lane<16 -> N=lane, {K0..7, K16..23}; lane>=16 -> same N, {K8..15, K24..31}
    int ncol = lane & 15, ko = (lane >> 4) * 8;
    for (int ni = 0; ni < 2; ++ni) {
      const h16* p = &sB[cur][(nw + ni * 16 + ncol) * 32 + ko];
      v8h lo = *(const v8h*)&p[0];
      v8h hi = *(const v8h*)&p[16];
      bf[ni] = __builtin_shufflevector(lo, hi, 0, 1, 2, 3, 4, 5, 6, 7,
                                       8, 9, 10, 11, 12, 13, 14, 15);
    }
    for (int mi = 0; mi < 4; ++mi)
      for (int ni = 0; ni < 2; ++ni)
        acc[mi][ni] = __builtin_amdgcn_wmma_f32_16x16x32_f16(
            false, af[mi], false, bf[ni], (short)0, acc[mi][ni], false, false);

    // 3) commit staged registers into the other buffer
    if (kt + 1 < nK) {
      for (int j = 0; j < 4; ++j)
        *(v8h*)&sA[nxt][ar[j] * 32 + ac[j]] = aReg[j];
      *(v8h*)&sB[nxt][br * 32 + bc] = bReg;
    }
    __syncthreads();
  }

  // epilogue: C/D layout -> lane&15 = N, vgpr v + 8*(lane>>4) = M
  int ncl = lane & 15, mof = (lane >> 4) * 8;
  for (int ni = 0; ni < 2; ++ni) {
    int gn = n0 + nw + ni * 16 + ncl;
    float bv = bias[gn];
    for (int mi = 0; mi < 4; ++mi) {
      for (int v = 0; v < 8; ++v) {
        int gm = m0 + mw + mi * 16 + mof + v;
        float val = acc[mi][ni][v] + bv;
        if (act) val = mishf(val);
        if (outF) outF[(size_t)gm * N + gn] = val;
        if (outH) outH[(size_t)gm * N + gn] = (h16)val;
      }
    }
  }
}

// ============================================================
// K6: trig precompute for reconstruction
// grid: B blocks, 256 threads
// ============================================================
__global__ void trig_kernel(const float* __restrict__ utv,
                            const float* __restrict__ btv,
                            float* cb, float* sb, float* cu, float* su) {
  int b = blockIdx.x, t = threadIdx.x;  // 256
  float th = btv[(size_t)b * 256 + t];
  cb[(size_t)b * 256 + t] = cosf(th) * 0.125f;  // 1/sqrt(64)
  sb[(size_t)b * 256 + t] = sinf(th) * 0.125f;
  if (t < 64) {
    float tu = utv[(size_t)b * 64 + t];
    cu[(size_t)b * 64 + t] = cosf(tu) * 0.25f;  // 1/sqrt(16)
    su[(size_t)b * 64 + t] = sinf(tu) * 0.25f;
  }
}

// ============================================================
// K7: reconstruction -> out (B,SC,DS,DS,2); memory-bound on H
// grid: B*SC blocks, 128 threads
// ============================================================
__global__ void recon_kernel(const float* __restrict__ Hr,
                             const float* __restrict__ Hi,
                             const float* __restrict__ cb,
                             const float* __restrict__ sb,
                             const float* __restrict__ cu,
                             const float* __restrict__ su,
                             const float* __restrict__ dig0,
                             const float* __restrict__ dig1,
                             const float* __restrict__ dig2,
                             const float* __restrict__ dig3,
                             float* __restrict__ outp) {
  int blk = blockIdx.x;
  int b = blk >> 6, sc = blk & 63;
  int t = threadIdx.x;  // 128
  __shared__ float sFr[128], sFi[128];  // [bs*2+d]
  __shared__ float sWr[32], sWi[32];    // [d*16+u]
  __shared__ float str[32], sti[32];    // [u*2+d]
  __shared__ float sHr[kUE * kBS], sHi[kUE * kBS];
  __shared__ float red[128];

  size_t dbase = (size_t)b * 512 + sc * 8;

  // ---- F2 = exp(i*bt)/8 @ FBB, then Frobenius-normalize ----
  float fr, fi;
  {
    int bs = t >> 1, d = t & 1;
    fr = 0.f; fi = 0.f;
    for (int r = 0; r < kRF; ++r) {
      float c = cb[(size_t)b * 256 + bs * 4 + r];
      float s = sb[(size_t)b * 256 + bs * 4 + r];
      float br = dig2[dbase + r * 2 + d];  // FBBr (B,SC,RF,DS)
      float bi = dig3[dbase + r * 2 + d];
      fr += c * br - s * bi;
      fi += c * bi + s * br;
    }
  }
  red[t] = fr * fr + fi * fi;
  __syncthreads();
  for (int s = 64; s > 0; s >>= 1) { if (t < s) red[t] += red[t + s]; __syncthreads(); }
  {
    float invp = 1.0f / sqrtf(red[0]);
    sFr[t] = fr * invp;
    sFi[t] = fi * invp;
  }
  __syncthreads();

  // ---- W2 = WBB @ exp(i*ut)/4, normalize * sqrt(2) ----
  float wr = 0.f, wi = 0.f;
  {
    int d = t >> 4, u = t & 15;
    if (t < 32) {
      for (int r = 0; r < kRF; ++r) {
        float br = dig0[dbase + d * 4 + r];  // WBBr (B,SC,DS,RF)
        float bi = dig1[dbase + d * 4 + r];
        float c = cu[(size_t)b * 64 + r * 16 + u];
        float s = su[(size_t)b * 64 + r * 16 + u];
        wr += br * c - bi * s;
        wi += br * s + bi * c;
      }
    }
  }
  red[t] = (t < 32) ? (wr * wr + wi * wi) : 0.0f;
  __syncthreads();
  for (int s = 64; s > 0; s >>= 1) { if (t < s) red[t] += red[t + s]; __syncthreads(); }
  if (t < 32) {
    float sca = 1.41421356237f / sqrtf(red[0]);
    sWr[t] = wr * sca;
    sWi[t] = wi * sca;
  }

  // ---- stage H tile ----
  size_t hb = ((size_t)(b * kSC + sc)) * kUE * kBS;
  for (int idx = t; idx < kUE * kBS; idx += 128) {
    sHr[idx] = Hr[hb + idx];
    sHi[idx] = Hi[hb + idx];
  }
  __syncthreads();

  // ---- t2 = H @ F2 ----
  if (t < 32) {
    int u = t >> 1, d = t & 1;
    float tr = 0.f, ti = 0.f;
    for (int k = 0; k < kBS; ++k) {
      float a = sHr[u * kBS + k], bq = sHi[u * kBS + k];
      float c = sFr[k * 2 + d], dd = sFi[k * 2 + d];
      tr += a * c - bq * dd;
      ti += a * dd + bq * c;
    }
    str[t] = tr;
    sti[t] = ti;
  }
  __syncthreads();

  // ---- out = W2 @ t2, stacked (real, imag) last dim ----
  if (t < 8) {
    int d1 = (t >> 2) & 1, d2 = (t >> 1) & 1, part = t & 1;
    float v = 0.f;
    for (int u = 0; u < kUE; ++u) {
      float a = sWr[d1 * 16 + u], bq = sWi[d1 * 16 + u];
      float c = str[u * 2 + d2], dd = sti[u * 2 + d2];
      v += part ? (a * dd + bq * c) : (a * c - bq * dd);
    }
    outp[(((size_t)(b * kSC + sc) * 2 + d1) * 2 + d2) * 2 + part] = v;
  }
}

// ============================================================
// host orchestration
// ============================================================
extern "C" void kernel_launch(void* const* d_in, const int* in_sizes, int n_in,
                              void* d_out, int out_size, void* d_ws, size_t ws_size,
                              hipStream_t stream) {
  (void)in_sizes; (void)n_in; (void)ws_size;
  const float* Hr    = (const float*)d_in[0];
  const float* Hi    = (const float*)d_in[1];
  const float* WRFt  = (const float*)d_in[2];
  const float* FRFt  = (const float*)d_in[3];
  const float* WBBr  = (const float*)d_in[4];
  const float* WBBi  = (const float*)d_in[5];
  const float* FBBr  = (const float*)d_in[6];
  const float* FBBi  = (const float*)d_in[7];
  const float* embed = (const float*)d_in[8];
  const float* uaW[4] = {(const float*)d_in[9],  (const float*)d_in[11],
                         (const float*)d_in[13], (const float*)d_in[15]};
  const float* uab[4] = {(const float*)d_in[10], (const float*)d_in[12],
                         (const float*)d_in[14], (const float*)d_in[16]};
  const float* baW[4] = {(const float*)d_in[17], (const float*)d_in[19],
                         (const float*)d_in[21], (const float*)d_in[23]};
  const float* bab[4] = {(const float*)d_in[18], (const float*)d_in[20],
                         (const float*)d_in[22], (const float*)d_in[24]};
  const float* dW[4]  = {(const float*)d_in[25], (const float*)d_in[27],
                         (const float*)d_in[29], (const float*)d_in[31]};
  const float* db[4]  = {(const float*)d_in[26], (const float*)d_in[28],
                         (const float*)d_in[30], (const float*)d_in[32]};
  float* outp = (float*)d_out;

  // ---- workspace carve-up (~40 MB total) ----
  char* wp = (char*)d_ws;
  auto alloc = [&](size_t bytes) -> void* {
    void* p = (void*)wp;
    wp += (bytes + 255) & ~(size_t)255;
    return p;
  };
  float* fvr   = (float*)alloc(64 * 64 * 4);
  float* fvi   = (float*)alloc(64 * 64 * 4);
  float* Wcr   = (float*)alloc(64 * 32 * 4);
  float* Wci   = (float*)alloc(64 * 32 * 4);
  float* fbb   = (float*)alloc((size_t)kB * kFEAT * 4);
  h16*   fbq16 = (h16*)  alloc((size_t)kB * kFEAT * 2);
  float* cbv   = (float*)alloc((size_t)kB * 256 * 4);
  float* sbv   = (float*)alloc((size_t)kB * 256 * 4);
  float* cuv   = (float*)alloc((size_t)kB * 64 * 4);
  float* suv   = (float*)alloc((size_t)kB * 64 * 4);
  float* utv   = (float*)alloc((size_t)kB * 64 * 4);
  float* btv   = (float*)alloc((size_t)kB * 256 * 4);
  float* dig   = (float*)alloc((size_t)4 * kB * 512 * 4);
  h16*   actA  = (h16*)  alloc((size_t)kB * 1024 * 2);
  h16*   actB  = (h16*)  alloc((size_t)kB * 1024 * 2);
  // layer shapes (K, N)
  const int uaK[4] = {256, 512, 1024, 512}, uaN[4] = {512, 1024, 512, 64};
  const int baK[4] = {256, 512, 1024, 512}, baN[4] = {512, 1024, 512, 256};
  const int dK[4]  = {256, 512, 1024, 1024}, dN[4] = {512, 1024, 1024, 512};
  h16* uaW16[4]; h16* baW16[4]; h16* dW16[4];
  for (int i = 0; i < 4; ++i) uaW16[i] = (h16*)alloc((size_t)uaK[i] * uaN[i] * 2);
  for (int i = 0; i < 4; ++i) baW16[i] = (h16*)alloc((size_t)baK[i] * baN[i] * 2);
  for (int i = 0; i < 4; ++i) dW16[i]  = (h16*)alloc((size_t)4 * dK[i] * dN[i] * 2);

  auto cvtT = [&](const float* s, h16* d, int Kd, int Nd) {
    dim3 g(Nd / 32, Kd / 32);
    cvt_t_kernel<<<g, 256, 0, stream>>>(s, d, Kd, Nd);
  };
  auto gemm = [&](const h16* A, const h16* Wm, const float* bias, float* oF,
                  h16* oH, int M, int N, int Kd, int act) {
    dim3 g(N / 64, M / 256);
    gemm_wmma_kernel<<<g, 256, 0, stream>>>(A, Wm, bias, oF, oH, M, N, Kd, act);
  };

  // ---- pipeline ----
  zero_loss_kernel<<<1, 32, 0, stream>>>(outp + out_size - 2);
  pilot_const_kernel<<<64, 64, 0, stream>>>(WRFt, FRFt, WBBr, WBBi, FBBr, FBBi,
                                            fvr, fvi, Wcr, Wci);
  pilot_proj_kernel<<<kB * kPS, 128, 0, stream>>>(Hr, Hi, fvr, fvi, Wcr, Wci, fbb);
  vq_kernel<<<kB, 128, 0, stream>>>(fbb, embed, fbq16, outp + out_size - 2);

  // weight transpose-converts: (K,N) f32 -> (N,K) f16
  for (int i = 0; i < 4; ++i) cvtT(uaW[i], uaW16[i], uaK[i], uaN[i]);
  for (int i = 0; i < 4; ++i) cvtT(baW[i], baW16[i], baK[i], baN[i]);
  for (int j = 0; j < 4; ++j)
    for (int i = 0; i < 4; ++i)
      cvtT(dW[j] + (size_t)i * dK[j] * dN[j],
           dW16[j] + (size_t)i * dK[j] * dN[j], dK[j], dN[j]);

  // ua chain: 256 -> 512 -> 1024 -> 512 -> 64
  gemm(fbq16, uaW16[0], uab[0], nullptr, actA, kB, 512, 256, 1);
  gemm(actA,  uaW16[1], uab[1], nullptr, actB, kB, 1024, 512, 1);
  gemm(actB,  uaW16[2], uab[2], nullptr, actA, kB, 512, 1024, 1);
  gemm(actA,  uaW16[3], uab[3], utv, nullptr,  kB, 64, 512, 0);
  // ba chain: 256 -> 512 -> 1024 -> 512 -> 256
  gemm(fbq16, baW16[0], bab[0], nullptr, actA, kB, 512, 256, 1);
  gemm(actA,  baW16[1], bab[1], nullptr, actB, kB, 1024, 512, 1);
  gemm(actB,  baW16[2], bab[2], nullptr, actA, kB, 512, 1024, 1);
  gemm(actA,  baW16[3], bab[3], btv, nullptr,  kB, 256, 512, 0);
  // digital chains i=0..3: 256 -> 512 -> 1024 -> 1024 -> 512
  for (int i = 0; i < 4; ++i) {
    gemm(fbq16, dW16[0] + (size_t)i * 256 * 512,   db[0] + i * 512,  nullptr, actA,
         kB, 512, 256, 1);
    gemm(actA,  dW16[1] + (size_t)i * 512 * 1024,  db[1] + i * 1024, nullptr, actB,
         kB, 1024, 512, 1);
    gemm(actB,  dW16[2] + (size_t)i * 1024 * 1024, db[2] + i * 1024, nullptr, actA,
         kB, 1024, 1024, 1);
    gemm(actA,  dW16[3] + (size_t)i * 1024 * 512,  db[3] + i * 512,
         dig + (size_t)i * kB * 512, nullptr, kB, 512, 1024, 0);
  }

  trig_kernel<<<kB, 256, 0, stream>>>(utv, btv, cbv, sbv, cuv, suv);
  recon_kernel<<<kB * kSC, 128, 0, stream>>>(
      Hr, Hi, cbv, sbv, cuv, suv,
      dig + (size_t)0 * kB * 512, dig + (size_t)1 * kB * 512,
      dig + (size_t)2 * kB * 512, dig + (size_t)3 * kB * 512, outp);
}
